// Net_18820546691331
// MI455X (gfx1250) — matrix operand order repaired and checked by reading, not compiled, not run on hardware
//
#include <hip/hip_runtime.h>
#include <hip/hip_bf16.h>

typedef __attribute__((ext_vector_type(16))) __bf16 v16bf;
typedef __attribute__((ext_vector_type(8)))  float  v8f;

#define D_FEAT 1024
#define NSP    16
#define NPAIR  120          // 16*15/2
#define MROWS  240          // 2*NPAIR

// ---------------------------------------------------------------------------
// Transpose fp32 W[K][N] -> bf16 WT[N][K] (coalesced both ways via LDS tile)
// ---------------------------------------------------------------------------
__global__ __launch_bounds__(256) void transpose_bf16(
    const float* __restrict__ W, __bf16* __restrict__ WT, int K, int N)
{
    __shared__ float tile[32][33];
    const int tx = threadIdx.x & 31;
    const int ty = threadIdx.x >> 5;          // 0..7
    const int bn = blockIdx.x * 32;           // N tile
    const int bk = blockIdx.y * 32;           // K tile
#pragma unroll
    for (int i = 0; i < 4; ++i) {
        int k = bk + ty + i * 8;
        tile[ty + i * 8][tx] = W[(size_t)k * N + bn + tx];
    }
    __syncthreads();
#pragma unroll
    for (int i = 0; i < 4; ++i) {
        int n = bn + ty + i * 8;
        WT[(size_t)n * K + bk + tx] = (__bf16)tile[tx][ty + i * 8];
    }
}

// ---------------------------------------------------------------------------
// Build X (MROWS x 2048, bf16): row p       = [h_i, h_j]
//                               row 120+p   = [h_j, h_i]
// ---------------------------------------------------------------------------
__global__ __launch_bounds__(256) void build_x(
    const float* __restrict__ h, __bf16* __restrict__ X, int off)
{
    int t = blockIdx.x * 256 + threadIdx.x;   // MROWS*2048 threads
    int row = t >> 11;
    int col = t & 2047;
    int pr = row, swap = 0;
    if (row >= NPAIR) { pr = row - NPAIR; swap = 1; }
    // pair index -> (i, j), i < j, row-major over upper triangle of 16x16
    int idx = pr, i = 0, rem = NSP - 1;
    while (idx >= rem) { idx -= rem; --rem; ++i; }
    int j = i + 1 + idx;
    int first  = swap ? j : i;
    int second = swap ? i : j;
    int sr = (col < D_FEAT) ? first : second;
    int sc = col & (D_FEAT - 1);
    X[t] = (__bf16)h[(size_t)(off + sr) * D_FEAT + sc];
}

// ---------------------------------------------------------------------------
// WMMA GEMM: C[M x N] = act(A[M x K] @ W[K x N] + bias), A bf16 row-major,
// W given as WT[N x K] bf16 row-major. Wave = one 16-col tile x 5 row tiles.
// Grid: x = N/128 (8 waves/block), y = 3 (each y covers 80 rows = 5 tiles).
// ---------------------------------------------------------------------------
template<bool SILU, bool OUTBF16>
__global__ __launch_bounds__(256) void gemm_wmma(
    const __bf16* __restrict__ A, const __bf16* __restrict__ WT,
    const float* __restrict__ bias, void* __restrict__ Cout, int K, int N)
{
    const int lane = threadIdx.x & 31;
    const int wv   = threadIdx.x >> 5;                 // wave in block (0..7)
    const int n0   = (blockIdx.x * 8 + wv) * 16;       // N tile origin
    const int mg   = blockIdx.y * 80;                  // 5 M-tiles of 16
    const int half = lane >> 4;                        // 0|1 (K half select)
    const int l15  = lane & 15;

    const __bf16* ap[5];
#pragma unroll
    for (int t = 0; t < 5; ++t)
        ap[t] = A + (size_t)(mg + t * 16 + l15) * K + half * 8;
    const __bf16* bp = WT + (size_t)(n0 + l15) * K + half * 8;

    v8f acc[5] = {};
    for (int k = 0; k < K; k += 32) {
        union { v16bf v; uint4 q[2]; } bu;
        bu.q[0] = *(const uint4*)(bp + k);        // K = k+8h .. +7
        bu.q[1] = *(const uint4*)(bp + k + 16);   // K = k+16+8h .. +7
#pragma unroll
        for (int t = 0; t < 5; ++t) {
            union { v16bf v; uint4 q[2]; } au;
            au.q[0] = *(const uint4*)(ap[t] + k);        // K = k+8h .. +7
            au.q[1] = *(const uint4*)(ap[t] + k + 16);   // K = k+16+8h .. +7
            acc[t] = __builtin_amdgcn_wmma_f32_16x16x32_bf16(
                false, au.v, false, bu.v, (short)0, acc[t], false, false);
        }
    }

    const float bv = bias[n0 + l15];
#pragma unroll
    for (int t = 0; t < 5; ++t) {
#pragma unroll
        for (int r = 0; r < 8; ++r) {
            int m = mg + t * 16 + r + 8 * half;
            float v = acc[t][r] + bv;
            if (SILU) v = v * (1.0f / (1.0f + __expf(-v)));
            if (OUTBF16)
                ((__bf16*)Cout)[(size_t)m * N + n0 + l15] = (__bf16)v;
            else
                ((float*)Cout)[(size_t)m * N + n0 + l15] = v;
        }
    }
}

// ---------------------------------------------------------------------------
// a[c] = prod_p (H[p][c] - H[120+p][c]),  c in [0,2048)
// ---------------------------------------------------------------------------
__global__ __launch_bounds__(256) void prod_reduce(
    const float* __restrict__ H, float* __restrict__ a)
{
    int c = blockIdx.x * 256 + threadIdx.x;
    float p = 1.0f;
#pragma unroll 4
    for (int q = 0; q < NPAIR; ++q)
        p *= (H[(size_t)q * 2048 + c] - H[(size_t)(q + NPAIR) * 2048 + c]);
    a[c] = p;
}

// ---------------------------------------------------------------------------
// t[j] = tanh(b[j] + sum_k a[k] * W[k][j])   (2048x2048 fp32, M=1)
// ---------------------------------------------------------------------------
__global__ __launch_bounds__(256) void mlp2_layer1(
    const float* __restrict__ a, const float* __restrict__ W,
    const float* __restrict__ b, float* __restrict__ t)
{
    int j = blockIdx.x * 256 + threadIdx.x;   // 2048 threads
    float s = b[j];
    for (int k = 0; k < 2048; ++k)
        s = fmaf(a[k], W[(size_t)k * 2048 + j], s);
    t[j] = tanhf(s);
}

// ---------------------------------------------------------------------------
// out (+)= log|sum_j t[j]*w2_1[j] + b2_1|   (spin 0 writes, spin 1 adds)
// ---------------------------------------------------------------------------
__global__ __launch_bounds__(256) void finalize(
    const float* __restrict__ t, const float* __restrict__ w,
    const float* __restrict__ b, float* __restrict__ out, int spin)
{
    __shared__ float red[256];
    int tid = threadIdx.x;
    float s = 0.0f;
    for (int j = tid; j < 2048; j += 256) s = fmaf(t[j], w[j], s);
    red[tid] = s;
    __syncthreads();
    for (int off = 128; off > 0; off >>= 1) {
        if (tid < off) red[tid] += red[tid + off];
        __syncthreads();
    }
    if (tid == 0) {
        float A = red[0] + b[0];
        float v = __logf(fabsf(A));
        if (spin == 0) out[0] = v;
        else           out[0] += v;
    }
}

// ---------------------------------------------------------------------------
extern "C" void kernel_launch(void* const* d_in, const int* in_sizes, int n_in,
                              void* d_out, int out_size, void* d_ws, size_t ws_size,
                              hipStream_t stream)
{
    const float* h_one = (const float*)d_in[0];
    const float* w1_0  = (const float*)d_in[1];
    const float* b1_0  = (const float*)d_in[2];
    const float* w1_1  = (const float*)d_in[3];
    const float* b1_1  = (const float*)d_in[4];
    const float* w1_2  = (const float*)d_in[5];
    const float* b1_2  = (const float*)d_in[6];
    const float* w2_0  = (const float*)d_in[7];
    const float* b2_0  = (const float*)d_in[8];
    const float* w2_1  = (const float*)d_in[9];
    const float* b2_1  = (const float*)d_in[10];
    float* out = (float*)d_out;

    // ---- workspace carve-up (bf16 weights + activations), 256B aligned ----
    char* ws = (char*)d_ws;
    size_t o = 0;
    auto carve = [&](size_t bytes) {
        void* p = ws + o;
        o = (o + bytes + 255) & ~(size_t)255;
        return p;
    };
    __bf16* wb0 = (__bf16*)carve((size_t)4096 * 2048 * 2);  // WT of w1_0 (N=4096,K=2048)
    __bf16* wb1 = (__bf16*)carve((size_t)4096 * 4096 * 2);  // WT of w1_1
    __bf16* wb2 = (__bf16*)carve((size_t)2048 * 4096 * 2);  // WT of w1_2 (N=2048,K=4096)
    __bf16* X   = (__bf16*)carve((size_t)MROWS * 2048 * 2);
    __bf16* Y1  = (__bf16*)carve((size_t)MROWS * 4096 * 2);
    __bf16* Y2  = (__bf16*)carve((size_t)MROWS * 4096 * 2);
    float*  H   = (float*) carve((size_t)MROWS * 2048 * 4);
    float*  av  = (float*) carve((size_t)2048 * 4);
    float*  tv  = (float*) carve((size_t)2048 * 4);
    (void)ws_size; (void)in_sizes; (void)n_in; (void)out_size;

    // ---- weight convert + transpose (fp32 K x N -> bf16 N x K) ----
    transpose_bf16<<<dim3(4096 / 32, 2048 / 32), 256, 0, stream>>>(w1_0, wb0, 2048, 4096);
    transpose_bf16<<<dim3(4096 / 32, 4096 / 32), 256, 0, stream>>>(w1_1, wb1, 4096, 4096);
    transpose_bf16<<<dim3(2048 / 32, 4096 / 32), 256, 0, stream>>>(w1_2, wb2, 4096, 2048);

    for (int spin = 0; spin < 2; ++spin) {
        const int off = spin * NSP;
        build_x<<<(MROWS * 2048) / 256, 256, 0, stream>>>(h_one, X, off);

        // mlp1: silu(X@W0+b0) -> silu(@W1+b1) -> @W2+b2
        gemm_wmma<true,  true ><<<dim3(32, 3), 256, 0, stream>>>(X,  wb0, b1_0, Y1, 2048, 4096);
        gemm_wmma<true,  true ><<<dim3(32, 3), 256, 0, stream>>>(Y1, wb1, b1_1, Y2, 4096, 4096);
        gemm_wmma<false, false><<<dim3(16, 3), 256, 0, stream>>>(Y2, wb2, b1_2, H,  4096, 2048);

        // antisymmetrize + product over pairs
        prod_reduce<<<2048 / 256, 256, 0, stream>>>(H, av);

        // mlp2 (M = 1) + log|A| accumulation
        mlp2_layer1<<<2048 / 256, 256, 0, stream>>>(av, w2_0, b2_0, tv);
        finalize<<<1, 256, 0, stream>>>(tv, w2_1, b2_1, out, spin);
    }
}
